// GCNEncoder_89773406421559
// MI455X (gfx1250) — matrix operand order repaired and checked by reading, not compiled, not run on hardware
//
#include <hip/hip_runtime.h>

typedef __attribute__((ext_vector_type(16))) _Float16 v16h;
typedef __attribute__((ext_vector_type(8)))  _Float16 v8h;
typedef __attribute__((ext_vector_type(8)))  float    v8f;

static constexpr int IN_DIM = 256;
static constexpr int HID    = 128;
static constexpr int ZDIM   = 64;

static inline unsigned cdiv(long long a, long long b) { return (unsigned)((a + b - 1) / b); }

// ---------------------------------------------------------------------------
// Degree with self loops: deg = 1 + count(dst); dinv = rsqrt(deg)
// ---------------------------------------------------------------------------
__global__ void gcn_init_deg(float* __restrict__ deg, int n) {
    int i = blockIdx.x * blockDim.x + threadIdx.x;
    if (i < n) deg[i] = 1.0f;
}

__global__ void gcn_edge_deg(const int* __restrict__ ei, float* __restrict__ deg, int e) {
    int i = blockIdx.x * blockDim.x + threadIdx.x;
    if (i < e) atomicAdd(&deg[ei[(size_t)e + i]], 1.0f);  // dst half
}

__global__ void gcn_rsqrt_inplace(float* __restrict__ d, int n) {
    int i = blockIdx.x * blockDim.x + threadIdx.x;
    if (i < n) d[i] = rsqrtf(d[i]);
}

// ---------------------------------------------------------------------------
// Weight convert + transpose: Wt[o][k] = (f16) W[k][o]   (B-fragment friendly)
// ---------------------------------------------------------------------------
__global__ void gcn_cvt_wt(const float* __restrict__ W, _Float16* __restrict__ Wt,
                           int K, int O) {
    int idx = blockIdx.x * blockDim.x + threadIdx.x;
    if (idx >= K * O) return;
    int k = idx / O, o = idx - k * O;
    Wt[(size_t)o * K + k] = (_Float16)W[(size_t)k * O + o];
}

// ---------------------------------------------------------------------------
// LayerNorm over 256 features, one wave32 per row, output f16
// ---------------------------------------------------------------------------
__global__ void gcn_layernorm_f16(const float* __restrict__ x,
                                  const float* __restrict__ gamma,
                                  const float* __restrict__ beta,
                                  _Float16* __restrict__ h0, int n) {
    int wid  = (blockIdx.x * blockDim.x + threadIdx.x) >> 5;
    int lane = threadIdx.x & 31;
    if (wid >= n) return;
    const float* xr = x + (size_t)wid * IN_DIM;

    float v[8];
    #pragma unroll
    for (int i = 0; i < 8; ++i) v[i] = xr[lane * 8 + i];

    float s = 0.f, s2 = 0.f;
    #pragma unroll
    for (int i = 0; i < 8; ++i) { s += v[i]; s2 += v[i] * v[i]; }
    #pragma unroll
    for (int off = 16; off > 0; off >>= 1) {
        s  += __shfl_xor(s,  off, 32);
        s2 += __shfl_xor(s2, off, 32);
    }
    float mu  = s * (1.0f / IN_DIM);
    float var = s2 * (1.0f / IN_DIM) - mu * mu;
    float r   = rsqrtf(var + 1e-5f);

    v8h o;
    #pragma unroll
    for (int i = 0; i < 8; ++i) {
        int j = lane * 8 + i;
        o[i] = (_Float16)((v[i] - mu) * r * gamma[j] + beta[j]);
    }
    *reinterpret_cast<v8h*>(h0 + (size_t)wid * IN_DIM + lane * 8) = o;
}

// ---------------------------------------------------------------------------
// WMMA GEMM: C(nrows x DOUT, f32) = A(nrows x K, f16) * Wt^T (Wt is DOUT x K).
// The whole weight matrix is staged once per block into LDS in *fragment
// order*: chunk ((nt*KT + kt)*32 + lane) holds the 16 halves lane `lane`
// needs for WMMA fragment (kt, nt). Inner-loop B reads become linear
// conflict-free ds_load_b128 pairs that co-execute with the WMMA pipe.
// One wave computes a 16-row stripe x full DOUT (register-blocked n-tiles).
// ---------------------------------------------------------------------------
template<int K, int DOUT>
__global__ __launch_bounds__(256)
void gcn_gemm_wmma(const _Float16* __restrict__ A,
                   const _Float16* __restrict__ Wt,
                   float* __restrict__ C, int nrows) {
    constexpr int NT = DOUT / 16;
    constexpr int KT = K / 32;
    constexpr int CHUNKS = NT * KT * 32;          // 16-half chunks in B

    __shared__ _Float16 Bs[DOUT * K];             // <= 64 KB (layer 1)

    // --- cooperative stage: global (row-major Wt) -> LDS (fragment order) ---
    for (int c = threadIdx.x; c < CHUNKS; c += 256) {
        int lc   = c & 31;                        // lane this chunk feeds
        int t    = c >> 5;                        // nt*KT + kt
        int kt   = t % KT;
        int nt   = t / KT;
        int row  = nt * 16 + (lc & 15);           // N index
        int koff = kt * 32 + (lc >> 4) * 16;      // K offset
        const uint4* src = reinterpret_cast<const uint4*>(Wt + (size_t)row * K + koff);
        uint4* dst = reinterpret_cast<uint4*>(Bs + (size_t)c * 16);
        dst[0] = src[0];
        dst[1] = src[1];
    }
    __syncthreads();

    const int lane = threadIdx.x & 31;
    const int rb   = (blockIdx.x * blockDim.x + threadIdx.x) >> 5;
    if (rb * 16 >= nrows) return;                 // whole-wave uniform exit
    const int lo = lane & 15;
    const int hi = lane >> 4;

    v8f acc[NT] = {};
    const _Float16* arow = A + (size_t)(rb * 16 + lo) * K;

    #pragma unroll
    for (int kt = 0; kt < KT; ++kt) {
        // A fragment: M = lane&15; halves 0..7 -> K = kt*32 + hi*16 .. +8
        //                          halves 8..15 -> same + 8
        v8h a0 = *reinterpret_cast<const v8h*>(arow + kt * 32 + hi * 16);
        v8h a1 = *reinterpret_cast<const v8h*>(arow + kt * 32 + hi * 16 + 8);
        v16h a;
        #pragma unroll
        for (int i = 0; i < 8; ++i) { a[i] = a0[i]; a[i + 8] = a1[i]; }

        #pragma unroll
        for (int nt = 0; nt < NT; ++nt) {
            const _Float16* bp = Bs + (((size_t)nt * KT + kt) * 32 + lane) * 16;
            v8h b0 = *reinterpret_cast<const v8h*>(bp);
            v8h b1 = *reinterpret_cast<const v8h*>(bp + 8);
            v16h b;
            #pragma unroll
            for (int i = 0; i < 8; ++i) { b[i] = b0[i]; b[i + 8] = b1[i]; }

            acc[nt] = __builtin_amdgcn_wmma_f32_16x16x32_f16(
                false, a, false, b, (short)0, acc[nt], false, false);
        }
    }

    #pragma unroll
    for (int nt = 0; nt < NT; ++nt) {
        #pragma unroll
        for (int i = 0; i < 8; ++i) {
            int row = rb * 16 + hi * 8 + i;
            C[(size_t)row * DOUT + nt * 16 + lo] = acc[nt][i];
        }
    }
}

// ---------------------------------------------------------------------------
// agg[i,:] = hW[i,:] * dinv[i]^2   (self-loop term initializes accumulator)
// ---------------------------------------------------------------------------
template<int D>
__global__ void gcn_self_init(const float* __restrict__ hW,
                              const float* __restrict__ dinv,
                              float* __restrict__ agg, int n) {
    long long idx = (long long)blockIdx.x * blockDim.x + threadIdx.x;
    if (idx >= (long long)n * D) return;
    int i = (int)(idx / D);
    float di = dinv[i];
    agg[idx] = hW[idx] * di * di;
}

// ---------------------------------------------------------------------------
// Edge scatter: one wave32 per edge; agg[dst,:] += hW[src,:] * dinv[s]*dinv[d]
// (agg/hW are L2-resident at 25.6 MB, so these atomics resolve in L2)
// ---------------------------------------------------------------------------
template<int D>
__global__ void gcn_edge_agg(const int* __restrict__ ei,
                             const float* __restrict__ dinv,
                             const float* __restrict__ hW,
                             float* __restrict__ agg, int e) {
    int wid  = (blockIdx.x * blockDim.x + threadIdx.x) >> 5;
    int lane = threadIdx.x & 31;
    if (wid >= e) return;
    int s = ei[wid];
    int d = ei[(size_t)e + wid];
    float coef = dinv[s] * dinv[d];
    const float* hs = hW  + (size_t)s * D;
    float*       ad = agg + (size_t)d * D;
    #pragma unroll
    for (int k = 0; k < D / 32; ++k) {
        int j = lane + k * 32;
        atomicAdd(&ad[j], hs[j] * coef);
    }
}

// ---------------------------------------------------------------------------
// Epilogues: +bias (+ReLU) -> f16 for next layer, or f32 to output
// ---------------------------------------------------------------------------
template<int D>
__global__ void gcn_bias_relu_f16(const float* __restrict__ agg,
                                  const float* __restrict__ bias,
                                  _Float16* __restrict__ out, int n) {
    long long idx = (long long)blockIdx.x * blockDim.x + threadIdx.x;
    if (idx >= (long long)n * D) return;
    int j = (int)(idx % D);
    float v = agg[idx] + bias[j];
    out[idx] = (_Float16)fmaxf(v, 0.0f);
}

template<int D>
__global__ void gcn_bias_f32(const float* __restrict__ agg,
                             const float* __restrict__ bias,
                             float* __restrict__ out, int n) {
    long long idx = (long long)blockIdx.x * blockDim.x + threadIdx.x;
    if (idx >= (long long)n * D) return;
    int j = (int)(idx % D);
    out[idx] = agg[idx] + bias[j];
}

// ---------------------------------------------------------------------------
// Host launcher
// ---------------------------------------------------------------------------
extern "C" void kernel_launch(void* const* d_in, const int* in_sizes, int n_in,
                              void* d_out, int out_size, void* d_ws, size_t ws_size,
                              hipStream_t stream) {
    const float* x     = (const float*)d_in[0];
    const int*   ei    = (const int*)  d_in[1];   // (2,E): [0,E)=src, [E,2E)=dst
    const float* gamma = (const float*)d_in[2];
    const float* beta  = (const float*)d_in[3];
    const float* W1    = (const float*)d_in[4];
    const float* b1    = (const float*)d_in[5];
    const float* W2    = (const float*)d_in[6];
    const float* b2    = (const float*)d_in[7];
    const float* W3    = (const float*)d_in[8];
    const float* b3    = (const float*)d_in[9];

    const int N = in_sizes[0] / IN_DIM;
    const int E = in_sizes[1] / 2;

    // Workspace carve-out (256B aligned slices), ~103 MB total
    char* ws = (char*)d_ws;
    size_t off = 0;
    auto take = [&](size_t bytes) -> char* {
        char* p = ws + off;
        off += (bytes + 255) & ~(size_t)255;
        return p;
    };
    float*    dinv = (float*)   take((size_t)N * 4);          // deg -> dinv in place
    _Float16* h0   = (_Float16*)take((size_t)N * IN_DIM * 2);
    _Float16* h1   = (_Float16*)take((size_t)N * HID * 2);
    _Float16* h2   = (_Float16*)take((size_t)N * HID * 2);
    float*    hW   = (float*)   take((size_t)N * HID * 4);
    float*    agg  = (float*)   take((size_t)N * HID * 4);
    _Float16* Wt1  = (_Float16*)take((size_t)IN_DIM * HID * 2);
    _Float16* Wt2  = (_Float16*)take((size_t)HID * HID * 2);
    _Float16* Wt3  = (_Float16*)take((size_t)HID * ZDIM * 2);

    const int T = 256;

    // degrees
    gcn_init_deg<<<cdiv(N, T), T, 0, stream>>>(dinv, N);
    gcn_edge_deg<<<cdiv(E, T), T, 0, stream>>>(ei, dinv, E);
    gcn_rsqrt_inplace<<<cdiv(N, T), T, 0, stream>>>(dinv, N);

    // weights -> f16 transposed
    gcn_cvt_wt<<<cdiv(IN_DIM * HID, T), T, 0, stream>>>(W1, Wt1, IN_DIM, HID);
    gcn_cvt_wt<<<cdiv(HID * HID,   T), T, 0, stream>>>(W2, Wt2, HID, HID);
    gcn_cvt_wt<<<cdiv(HID * ZDIM,  T), T, 0, stream>>>(W3, Wt3, HID, ZDIM);

    // layernorm -> h0 (f16)
    gcn_layernorm_f16<<<cdiv((long long)N * 32, T), T, 0, stream>>>(x, gamma, beta, h0, N);

    const unsigned gemm_grid = cdiv((long long)(N / 16) * 32, T);

    // ----- layer 1: 256 -> 128, ReLU -----
    gcn_gemm_wmma<IN_DIM, HID><<<gemm_grid, T, 0, stream>>>(h0, Wt1, hW, N);
    gcn_self_init<HID><<<cdiv((long long)N * HID, T), T, 0, stream>>>(hW, dinv, agg, N);
    gcn_edge_agg<HID><<<cdiv((long long)E * 32, T), T, 0, stream>>>(ei, dinv, hW, agg, E);
    gcn_bias_relu_f16<HID><<<cdiv((long long)N * HID, T), T, 0, stream>>>(agg, b1, h1, N);

    // ----- layer 2: 128 -> 128, ReLU -----
    gcn_gemm_wmma<HID, HID><<<gemm_grid, T, 0, stream>>>(h1, Wt2, hW, N);
    gcn_self_init<HID><<<cdiv((long long)N * HID, T), T, 0, stream>>>(hW, dinv, agg, N);
    gcn_edge_agg<HID><<<cdiv((long long)E * 32, T), T, 0, stream>>>(ei, dinv, hW, agg, E);
    gcn_bias_relu_f16<HID><<<cdiv((long long)N * HID, T), T, 0, stream>>>(agg, b2, h2, N);

    // ----- layer 3: 128 -> 64, linear, f32 output -----
    gcn_gemm_wmma<HID, ZDIM><<<gemm_grid, T, 0, stream>>>(h2, Wt3, hW, N);
    gcn_self_init<ZDIM><<<cdiv((long long)N * ZDIM, T), T, 0, stream>>>(hW, dinv, agg, N);
    gcn_edge_agg<ZDIM><<<cdiv((long long)E * 32, T), T, 0, stream>>>(ei, dinv, hW, agg, E);
    gcn_bias_f32<ZDIM><<<cdiv((long long)N * ZDIM, T), T, 0, stream>>>(agg, b3, (float*)d_out, N);
}
